// Wavenet_44933947851073
// MI455X (gfx1250) — matrix operand order repaired and checked by reading, not compiled, not run on hardware
//
#include <hip/hip_runtime.h>
#include <hip/hip_bf16.h>
#include <stdint.h>

// ---------------------------------------------------------------------------
// WaveNet on MI455X (gfx1250), bf16 WMMA + async-to-LDS staging.
//
// The nsynth batch-fold preserves the flat index p = t*2 + b, so every
// dilated conv (dilation d) is a 3-tap conv with stride 2*d over a single
// [8192 pos][256 ch] activation matrix, zero-padded outside [0,8192).
// Activations are position-major so WMMA B-fragments are contiguous.
// B tiles are staged into LDS with GLOBAL_LOAD_ASYNC_TO_LDS_B128 (ASYNCcnt)
// and shared by the 4 waves of a block; padding handled at stage time.
// ---------------------------------------------------------------------------

typedef __attribute__((ext_vector_type(16))) __bf16 v16bf;
typedef __attribute__((ext_vector_type(8)))  __bf16 v8bf;
typedef __attribute__((ext_vector_type(8)))  float  v8f;

#define P_TOT 8192   // 4096 time * 2 batch interleaved: p = t*2 + b
#define CH    256
#define NL    30
#define BROW  264    // staged row stride in bf16 (256 + 8 pad): 528B = 33 quads -> bank-conflict-free

static __device__ __forceinline__ v8f vzero8() {
  v8f z;
#pragma unroll
  for (int i = 0; i < 8; ++i) z[i] = 0.0f;
  return z;
}

static __device__ __forceinline__ v8bf bzero8() {
  v8bf z;
#pragma unroll
  for (int i = 0; i < 8; ++i) z[i] = (__bf16)0.0f;
  return z;
}

static __device__ __forceinline__ v16bf pack16(v8bf lo, v8bf hi) {
  v16bf a;
#pragma unroll
  for (int i = 0; i < 8; ++i) { a[i] = lo[i]; a[8 + i] = hi[i]; }
  return a;
}

// A fragment (16x32 bf16), ISA layout: lane l holds row M = l&15, hl = l>>4.
//   element i in [0,8)  <-> K = hl*8 + i ; element i in [8,16) <-> K = 16 + hl*8 + (i-8)
static __device__ __forceinline__ v16bf load_a_frag(const __bf16* rowc0, int hl) {
  return pack16(*(const v8bf*)(rowc0 + hl * 8), *(const v8bf*)(rowc0 + 16 + hl * 8));
}

// B fragment (32x16 bf16): lane l holds column N = l&15, same K order as A fragment.
static __device__ __forceinline__ v16bf load_b_global(const __bf16* act, int q, int c0, int hl) {
  const __bf16* p = act + (size_t)q * CH + c0 + hl * 8;
  return pack16(*(const v8bf*)p, *(const v8bf*)(p + 16));
}

static __device__ __forceinline__ v16bf load_b_lds(const __bf16* stage, int row, int c0, int hl) {
  const __bf16* p = stage + row * BROW + c0 + hl * 8;   // 16B-aligned ds_load_b128 pair
  return pack16(*(const v8bf*)p, *(const v8bf*)(p + 16));
}

static __device__ __forceinline__ v8f wmma_bf16(v16bf a, v16bf b, v8f c) {
  return __builtin_amdgcn_wmma_f32_16x16x32_bf16(false, a, false, b, (short)0, c,
                                                 false, false);
}

// Cooperative async stage of 64 rows x 512B of `src` (positions qbase..qbase+63)
// into `stage` (row stride BROW), zero-filling out-of-range positions.
// 128 threads x 16 iters x 16B = 32KB.
static __device__ __forceinline__ void stage_tile_async(__bf16* stage,
                                                        const __bf16* src,
                                                        int qbase, int tid) {
#pragma unroll
  for (int j = 0; j < 16; ++j) {
    int id  = j * 128 + tid;       // 0..2047
    int row = id >> 5;             // 0..63
    int seg = id & 31;             // 16B segment within the 512B row
    int q = qbase + row;
    __bf16* lp = stage + row * BROW + seg * 8;
    if (q >= 0 && q < P_TOT) {
      unsigned ldsOff = (unsigned)((uintptr_t)lp - (uintptr_t)stage); // single __shared__ obj @ 0
      unsigned long long ga =
          (unsigned long long)(uintptr_t)(src + (size_t)q * CH + seg * 8);
      asm volatile("global_load_async_to_lds_b128 %0, %1, off"
                   :: "v"(ldsOff), "v"(ga) : "memory");
    } else {
      *(v8bf*)lp = bzero8();       // conv zero-padding
    }
  }
  asm volatile("s_wait_asynccnt 0x0" ::: "memory");
}

// ---------------------------------------------------------------------------
// Prep kernels (weights f32 -> bf16 repack; run every launch)
// ---------------------------------------------------------------------------
__global__ void prep_x_kernel(const float* __restrict__ x, float* __restrict__ X) {
  int p = blockIdx.x * blockDim.x + threadIdx.x;       // p = t*2 + b
  X[p] = x[(p & 1) * 4096 + (p >> 1)];
}

__global__ void cvt_f32_bf16_kernel(const float* __restrict__ src,
                                    __bf16* __restrict__ dst, int n) {
  int i = blockIdx.x * blockDim.x + threadIdx.x;
  if (i < n) dst[i] = (__bf16)src[i];
}

// gate/feat: [k][cout][cin][tap] f32 -> [k][tap][cout][cin] bf16 (cin contiguous)
__global__ void prep_gf_kernel(const float* __restrict__ src,
                               __bf16* __restrict__ dst, int n) {
  int o = blockIdx.x * blockDim.x + threadIdx.x;
  if (o >= n) return;
  int cin  = o & 255;
  int cout = (o >> 8) & 255;
  int kt   = o >> 16;            // k*3 + tap
  int tap  = kt % 3;
  int k    = kt / 3;
  dst[o] = (__bf16)src[(((size_t)(k * 256 + cout) * 256 + cin)) * 3 + tap];
}

// init conv: 1 -> 256 channels, K=3; tiny, plain VALU. Writes H[p][r] bf16.
__global__ void init_kernel(const float* __restrict__ X,
                            const float* __restrict__ w,   // [256][3]
                            const float* __restrict__ bias,
                            __bf16* __restrict__ H) {
  int i = blockIdx.x * blockDim.x + threadIdx.x;  // i = p*256 + r
  int r = i & 255;
  int p = i >> 8;
  float acc = bias[r];
#pragma unroll
  for (int tap = 0; tap < 3; ++tap) {
    int q = p + (tap - 1) * 2;
    if (q >= 0 && q < P_TOT) acc += w[r * 3 + tap] * X[q];
  }
  H[i] = (__bf16)acc;
}

// ---------------------------------------------------------------------------
// Fused gate+feat GEMM (3 taps, stride=step) + sigmoid*tanh -> RES (bf16)
// Block: 4 waves sharing one 64-position range; wave tile 16 cout x 64 pos.
// Per tap: async-stage H[p0+delta .. +63][0..255] to LDS, then 8 K-blocks.
// ---------------------------------------------------------------------------
__global__ __launch_bounds__(128)
void layer_gf_kernel(const __bf16* __restrict__ H,      // [P][256]
                     __bf16* __restrict__ RES,          // [P][256]
                     const __bf16* __restrict__ Wg,     // [3][256][256]
                     const __bf16* __restrict__ Wf,     // [3][256][256]
                     const float* __restrict__ gb,
                     const float* __restrict__ fb,
                     int step) {
  __shared__ __bf16 bstage[64 * BROW];
  const int tid  = threadIdx.x;
  const int lane = tid & 31;
  const int wv   = (blockIdx.x << 2) | (tid >> 5);
  const int cout0 = (wv & 15) * 16;     // 4 consecutive cout blocks per block
  const int p0    = (wv >> 4) * 64;     // shared by all 4 waves of the block
  const int hl = lane >> 4;
  const int nl = lane & 15;

  v8f accg[4], accf[4];
#pragma unroll
  for (int t = 0; t < 4; ++t) { accg[t] = vzero8(); accf[t] = vzero8(); }

  for (int tap = 0; tap < 3; ++tap) {
    const int delta = (tap - 1) * step;
    if (tap) __syncthreads();                       // previous tile fully consumed
    stage_tile_async(bstage, H, p0 + delta, tid);   // async -> LDS, s_wait_asynccnt
    __syncthreads();

    const __bf16* ga = Wg + (size_t)tap * 65536 + (size_t)(cout0 + nl) * 256;
    const __bf16* fa = Wf + (size_t)tap * 65536 + (size_t)(cout0 + nl) * 256;
    if (tap < 2) {   // global_prefetch_b8: next tap's weight rows
      __builtin_prefetch(ga + 65536, 0, 3);
      __builtin_prefetch(fa + 65536, 0, 3);
    }
#pragma unroll
    for (int cb = 0; cb < 8; ++cb) {
      const int c0 = cb * 32;
      v16bf ag = load_a_frag(ga + c0, hl);
      v16bf af = load_a_frag(fa + c0, hl);
#pragma unroll
      for (int t = 0; t < 4; ++t) {
        v16bf b = load_b_lds(bstage, t * 16 + nl, c0, hl);
        accg[t] = wmma_bf16(ag, b, accg[t]);
        accf[t] = wmma_bf16(af, b, accf[t]);
      }
    }
  }

  // D layout: element r <-> cout = cout0 + hl*8 + r  (contiguous 8-ch store)
#pragma unroll
  for (int t = 0; t < 4; ++t) {
    int p = p0 + t * 16 + nl;
    __bf16* dst = RES + (size_t)p * CH + cout0 + hl * 8;
    v8bf o;
#pragma unroll
    for (int r = 0; r < 8; ++r) {
      int co = cout0 + hl * 8 + r;
      float g = accg[t][r] + gb[co];
      float f = accf[t][r] + fb[co];
      float sg = 1.0f / (1.0f + __expf(-g));
      float e2 = __expf(2.0f * f);
      float th = (e2 - 1.0f) / (e2 + 1.0f);
      o[r] = (__bf16)(sg * th);
    }
    *(v8bf*)dst = o;
  }
}

// ---------------------------------------------------------------------------
// Fused 1x1 thru + skip GEMM:  h += Wt @ res ;  s += Ws @ res
// RES tile staged once to LDS (always in range), shared by 4 waves.
// ---------------------------------------------------------------------------
__global__ __launch_bounds__(128)
void layer_ts_kernel(const __bf16* __restrict__ RES,
                     __bf16* __restrict__ H,
                     float* __restrict__ S,             // [P][256] f32
                     const __bf16* __restrict__ Wt,     // [256][256]
                     const __bf16* __restrict__ Ws) {
  __shared__ __bf16 bstage[64 * BROW];
  const int tid  = threadIdx.x;
  const int lane = tid & 31;
  const int wv   = (blockIdx.x << 2) | (tid >> 5);
  const int cout0 = (wv & 15) * 16;
  const int p0    = (wv >> 4) * 64;
  const int hl = lane >> 4;
  const int nl = lane & 15;

  stage_tile_async(bstage, RES, p0, tid);
  __syncthreads();

  v8f acct[4], accs[4];
#pragma unroll
  for (int t = 0; t < 4; ++t) { acct[t] = vzero8(); accs[t] = vzero8(); }

#pragma unroll
  for (int cb = 0; cb < 8; ++cb) {
    const int c0 = cb * 32;
    v16bf at = load_a_frag(Wt + (size_t)(cout0 + nl) * 256 + c0, hl);
    v16bf as = load_a_frag(Ws + (size_t)(cout0 + nl) * 256 + c0, hl);
#pragma unroll
    for (int t = 0; t < 4; ++t) {
      v16bf b = load_b_lds(bstage, t * 16 + nl, c0, hl);
      acct[t] = wmma_bf16(at, b, acct[t]);
      accs[t] = wmma_bf16(as, b, accs[t]);
    }
  }

#pragma unroll
  for (int t = 0; t < 4; ++t) {
    int p = p0 + t * 16 + nl;
    __bf16* hp = H + (size_t)p * CH + cout0 + hl * 8;
    float*  sp = S + (size_t)p * CH + cout0 + hl * 8;
    v8bf hold = *(const v8bf*)hp;
    v8bf hnew;
#pragma unroll
    for (int r = 0; r < 8; ++r) hnew[r] = (__bf16)((float)hold[r] + acct[t][r]);
    *(v8bf*)hp = hnew;
#pragma unroll
    for (int r = 0; r < 8; ++r) sp[r] += accs[t][r];
  }
}

// ---------------------------------------------------------------------------
// iskip: S = Wis @ H + bias  (f32 out, overwrite). Positions always in range.
// ---------------------------------------------------------------------------
__global__ __launch_bounds__(128)
void iskip_kernel(const __bf16* __restrict__ H,
                  const __bf16* __restrict__ Wis,
                  const float* __restrict__ bias,
                  float* __restrict__ S) {
  const int lane = threadIdx.x & 31;
  const int wv   = (blockIdx.x << 2) | (threadIdx.x >> 5);
  const int cout0 = (wv & 15) * 16;
  const int p0    = (wv >> 4) * 64;
  const int hl = lane >> 4;
  const int nl = lane & 15;

  v8f acc[4];
#pragma unroll
  for (int t = 0; t < 4; ++t) acc[t] = vzero8();

#pragma unroll
  for (int cb = 0; cb < 8; ++cb) {
    const int c0 = cb * 32;
    v16bf a = load_a_frag(Wis + (size_t)(cout0 + nl) * 256 + c0, hl);
#pragma unroll
    for (int t = 0; t < 4; ++t) {
      v16bf b = load_b_global(H, p0 + t * 16 + nl, c0, hl);
      acc[t] = wmma_bf16(a, b, acc[t]);
    }
  }

#pragma unroll
  for (int t = 0; t < 4; ++t) {
    int p = p0 + t * 16 + nl;
    float* sp = S + (size_t)p * CH + cout0 + hl * 8;
#pragma unroll
    for (int r = 0; r < 8; ++r) sp[r] = acc[t][r] + bias[cout0 + hl * 8 + r];
  }
}

// ---------------------------------------------------------------------------
// final: out[b][co][t] = Wfi @ SB + bias, de-interleaving p = t*2 + b
// ---------------------------------------------------------------------------
__global__ __launch_bounds__(128)
void final_kernel(const __bf16* __restrict__ SB,
                  const __bf16* __restrict__ Wfi,
                  const float* __restrict__ bias,
                  float* __restrict__ out) {
  const int lane = threadIdx.x & 31;
  const int wv   = (blockIdx.x << 2) | (threadIdx.x >> 5);
  const int cout0 = (wv & 15) * 16;
  const int p0    = (wv >> 4) * 64;
  const int hl = lane >> 4;
  const int nl = lane & 15;

  v8f acc[4];
#pragma unroll
  for (int t = 0; t < 4; ++t) acc[t] = vzero8();

#pragma unroll
  for (int cb = 0; cb < 8; ++cb) {
    const int c0 = cb * 32;
    v16bf a = load_a_frag(Wfi + (size_t)(cout0 + nl) * 256 + c0, hl);
#pragma unroll
    for (int t = 0; t < 4; ++t) {
      v16bf b = load_b_global(SB, p0 + t * 16 + nl, c0, hl);
      acc[t] = wmma_bf16(a, b, acc[t]);
    }
  }

#pragma unroll
  for (int t = 0; t < 4; ++t) {
    int p  = p0 + t * 16 + nl;
    int tt = p >> 1;
    int bb = p & 1;
#pragma unroll
    for (int r = 0; r < 8; ++r) {
      int co = cout0 + hl * 8 + r;
      out[(size_t)bb * (256 * 4096) + (size_t)co * 4096 + tt] = acc[t][r] + bias[co];
    }
  }
}

// ---------------------------------------------------------------------------
extern "C" void kernel_launch(void* const* d_in, const int* in_sizes, int n_in,
                              void* d_out, int out_size, void* d_ws, size_t ws_size,
                              hipStream_t stream) {
  (void)in_sizes; (void)n_in; (void)out_size; (void)ws_size;
  const float* x       = (const float*)d_in[0];
  const float* init_w  = (const float*)d_in[1];
  const float* init_b  = (const float*)d_in[2];
  const float* iskip_w = (const float*)d_in[3];
  const float* iskip_b = (const float*)d_in[4];
  const float* gate_w  = (const float*)d_in[5];
  const float* gate_b  = (const float*)d_in[6];
  const float* feat_w  = (const float*)d_in[7];
  const float* feat_b  = (const float*)d_in[8];
  const float* skip_w  = (const float*)d_in[9];
  const float* thru_w  = (const float*)d_in[10];
  const float* final_w = (const float*)d_in[11];
  const float* final_b = (const float*)d_in[12];
  float* out = (float*)d_out;

  char* ws = (char*)d_ws;
  size_t off = 0;
  auto alloc = [&](size_t bytes) -> void* {
    void* p = ws + off;
    off = (off + bytes + 255) & ~(size_t)255;
    return p;
  };

  float*  X   = (float*) alloc((size_t)P_TOT * 4);
  __bf16* H   = (__bf16*)alloc((size_t)P_TOT * CH * 2);
  __bf16* RES = (__bf16*)alloc((size_t)P_TOT * CH * 2);
  float*  S   = (float*) alloc((size_t)P_TOT * CH * 4);
  __bf16* SB  = (__bf16*)alloc((size_t)P_TOT * CH * 2);
  __bf16* Wg  = (__bf16*)alloc((size_t)NL * 3 * 65536 * 2);
  __bf16* Wf  = (__bf16*)alloc((size_t)NL * 3 * 65536 * 2);
  __bf16* Wth = (__bf16*)alloc((size_t)NL * 65536 * 2);
  __bf16* Wsk = (__bf16*)alloc((size_t)NL * 65536 * 2);
  __bf16* Wis = (__bf16*)alloc((size_t)65536 * 2);
  __bf16* Wfi = (__bf16*)alloc((size_t)65536 * 2);
  // total ~53 MB of workspace

  // ---- weight / input repack (bf16) ----
  prep_x_kernel<<<P_TOT / 256, 256, 0, stream>>>(x, X);
  cvt_f32_bf16_kernel<<<(NL * 65536) / 256, 256, 0, stream>>>(thru_w, Wth, NL * 65536);
  cvt_f32_bf16_kernel<<<(NL * 65536) / 256, 256, 0, stream>>>(skip_w, Wsk, NL * 65536);
  cvt_f32_bf16_kernel<<<65536 / 256, 256, 0, stream>>>(iskip_w, Wis, 65536);
  cvt_f32_bf16_kernel<<<65536 / 256, 256, 0, stream>>>(final_w, Wfi, 65536);
  prep_gf_kernel<<<(NL * 3 * 65536) / 256, 256, 0, stream>>>(gate_w, Wg, NL * 3 * 65536);
  prep_gf_kernel<<<(NL * 3 * 65536) / 256, 256, 0, stream>>>(feat_w, Wf, NL * 3 * 65536);

  // ---- network ----
  init_kernel<<<(P_TOT * CH) / 256, 256, 0, stream>>>(X, init_w, init_b, H);
  iskip_kernel<<<512, 128, 0, stream>>>(H, Wis, iskip_b, S);

  for (int k = 0; k < NL; ++k) {
    int step = 2 << (k % 10);  // 2 * dilation
    layer_gf_kernel<<<512, 128, 0, stream>>>(
        H, RES, Wg + (size_t)k * 3 * 65536, Wf + (size_t)k * 3 * 65536,
        gate_b + k * 256, feat_b + k * 256, step);
    layer_ts_kernel<<<512, 128, 0, stream>>>(
        RES, H, S, Wth + (size_t)k * 65536, Wsk + (size_t)k * 65536);
  }

  cvt_f32_bf16_kernel<<<(P_TOT * CH) / 256, 256, 0, stream>>>(S, SB, P_TOT * CH);
  final_kernel<<<512, 128, 0, stream>>>(SB, Wfi, final_b, out);
}
